// SFTLanguageModel_70282844832022
// MI455X (gfx1250) — matrix operand order repaired and checked by reading, not compiled, not run on hardware
//
#include <hip/hip_runtime.h>
#include <math.h>

// ---------------------------------------------------------------------------
// Problem dims (match reference)
// ---------------------------------------------------------------------------
constexpr int Bz   = 8;
constexpr int Sq   = 512;
constexpr int Hd   = 768;
constexpr int Ly   = 12;
constexpr int NHd  = 12;
constexpr int Ff   = 3072;
constexpr int Vc   = 21128;
constexpr int Vpad = 21248;        // 166 * 128, zero-padded cls weight rows
constexpr int DHd  = Hd / NHd;     // 64
constexpr int Mtok = Bz * Sq;      // 4096
constexpr int SEP_ID = 102;

// ---------------------------------------------------------------------------
// WMMA types / helpers (CDNA5 gfx1250, wave32)
// ---------------------------------------------------------------------------
typedef __attribute__((ext_vector_type(16))) __bf16 v16bf;
typedef __attribute__((ext_vector_type(8)))  __bf16 v8bf;
typedef __attribute__((ext_vector_type(8)))  float  v8f;

__device__ __forceinline__ __bf16 f2bf(float f) {
  unsigned u = __builtin_bit_cast(unsigned, f);
  u += 0x7FFFu + ((u >> 16) & 1u);           // round-to-nearest-even
  unsigned short h = (unsigned short)(u >> 16);
  return __builtin_bit_cast(__bf16, h);
}

__device__ __forceinline__ v16bf cat8(v8bf lo, v8bf hi) {
  return __builtin_shufflevector(lo, hi, 0,1,2,3,4,5,6,7,8,9,10,11,12,13,14,15);
}

// A fragment (16x32 bf16): lane L -> row = L&15, khalf = L>>4.
// Row-major row pointer p (at k0): two contiguous 16B loads.
__device__ __forceinline__ v16bf load_a_frag(const __bf16* p, int half) {
  v8bf lo = *(const v8bf*)(p + half * 8);
  v8bf hi = *(const v8bf*)(p + 16 + half * 8);
  return cat8(lo, hi);
}

// B fragment (32x16 bf16): lane L -> col = L&15, khalf = L>>4 (same K order as A).
// Bt stored [n][k] row-major: two contiguous 16B loads at k0 + half*8 and k0 + 16 + half*8.
__device__ __forceinline__ v16bf load_b_frag(const __bf16* p, int half) {
  v8bf lo = *(const v8bf*)(p + half * 8);
  v8bf hi = *(const v8bf*)(p + 16 + half * 8);
  return cat8(lo, hi);
}

__device__ __forceinline__ v8f wmma_bf16(v16bf a, v16bf b, v8f c) {
  return __builtin_amdgcn_wmma_f32_16x16x32_bf16(false, a, false, b,
                                                 (short)0, c, false, false);
}

__device__ __forceinline__ float gelu_f(float x) {
  return 0.5f * x * (1.0f + erff(x * 0.70710678118654752f));
}

__device__ __forceinline__ float block_sum(float v, float* red) {
  int tid = threadIdx.x;
  red[tid] = v; __syncthreads();
  for (int s = 128; s > 0; s >>= 1) {
    if (tid < s) red[tid] += red[tid + s];
    __syncthreads();
  }
  float r = red[0]; __syncthreads();
  return r;
}

// ---------------------------------------------------------------------------
// SEP / mask metadata
// ---------------------------------------------------------------------------
__global__ void sep_kernel(const int* __restrict__ ids,
                           int* __restrict__ sepA, int* __restrict__ firstA) {
  int b = threadIdx.x;
  if (b >= Bz) return;
  int first = -1;
  for (int i = 0; i < Sq; ++i) {
    if (ids[b * Sq + i] == SEP_ID) { first = i; break; }
  }
  firstA[b] = first;
  sepA[b]   = (first < 0) ? -1 : (first < Sq - 2 ? first : Sq - 2);
}

// ---------------------------------------------------------------------------
// Weight convert + transpose (+ zero pad): W[K][N] f32 -> Wt[Npad][K] bf16
// block (32,8), grid (Npad/32, K/32)
// ---------------------------------------------------------------------------
__global__ __launch_bounds__(256) void wconv_kernel(
    const float* __restrict__ W, __bf16* __restrict__ Wt,
    int Kd, int Nd, int Npad) {
  __shared__ float t[32][33];
  int n0 = blockIdx.x * 32, k0 = blockIdx.y * 32;
  int tx = threadIdx.x, ty = threadIdx.y;
  #pragma unroll
  for (int yy = ty; yy < 32; yy += 8) {
    int k = k0 + yy, n = n0 + tx;
    t[yy][tx] = (n < Nd) ? W[(size_t)k * Nd + n] : 0.0f;
  }
  __syncthreads();
  #pragma unroll
  for (int yy = ty; yy < 32; yy += 8) {
    int n = n0 + yy, k = k0 + tx;
    if (n < Npad) Wt[(size_t)n * Kd + k] = f2bf(t[tx][yy]);
  }
}

// ---------------------------------------------------------------------------
// V transpose per (b,h): vbb[M][H] bf16 -> vt[(b*NH+h)*DH + d][Sq] bf16
// block (32,8), grid (DH/32, Sq/32, B*NH)
// ---------------------------------------------------------------------------
__global__ __launch_bounds__(256) void transpose_v_kernel(
    const __bf16* __restrict__ vbb, __bf16* __restrict__ vt) {
  __shared__ __bf16 t[32][33];
  int z = blockIdx.z, b = z / NHd, h = z % NHd;
  int d0 = blockIdx.x * 32, j0 = blockIdx.y * 32;
  int tx = threadIdx.x, ty = threadIdx.y;
  #pragma unroll
  for (int yy = ty; yy < 32; yy += 8)
    t[yy][tx] = vbb[((size_t)b * Sq + j0 + yy) * Hd + h * DHd + d0 + tx];
  __syncthreads();
  #pragma unroll
  for (int yy = ty; yy < 32; yy += 8)
    vt[((size_t)z * DHd + d0 + yy) * Sq + j0 + tx] = t[tx][yy];
}

// ---------------------------------------------------------------------------
// Embedding + LayerNorm -> f32 x and bf16 xb
// ---------------------------------------------------------------------------
__global__ __launch_bounds__(256) void embed_ln_kernel(
    const int* __restrict__ ids, const float* __restrict__ we,
    const float* __restrict__ pe, const float* __restrict__ te,
    const float* __restrict__ g, const float* __restrict__ bt,
    float* __restrict__ x, __bf16* __restrict__ xb) {
  __shared__ float red[256];
  int t = blockIdx.x, tid = threadIdx.x;
  int s = t & (Sq - 1);
  int id = ids[t];
  float v[3]; float ls = 0.f, lq = 0.f;
  #pragma unroll
  for (int e = 0; e < 3; ++e) {
    int hh = e * 256 + tid;
    float val = we[(size_t)id * Hd + hh] + pe[(size_t)s * Hd + hh] + te[hh];
    v[e] = val; ls += val; lq += val * val;
  }
  float sum = block_sum(ls, red);
  float sq  = block_sum(lq, red);
  float mean = sum / Hd;
  float var  = sq / Hd - mean * mean;
  float inv  = rsqrtf(var + 1e-12f);
  #pragma unroll
  for (int e = 0; e < 3; ++e) {
    int hh = e * 256 + tid;
    float o = (v[e] - mean) * inv * g[hh] + bt[hh];
    x[(size_t)t * Hd + hh]  = o;
    xb[(size_t)t * Hd + hh] = f2bf(o);
  }
}

// ---------------------------------------------------------------------------
// Residual + LayerNorm: out = LN(res + y)  -> f32 and bf16
// ---------------------------------------------------------------------------
__global__ __launch_bounds__(256) void add_res_ln_kernel(
    const float* __restrict__ res, const float* __restrict__ y,
    const float* __restrict__ g, const float* __restrict__ bt,
    float* __restrict__ out, __bf16* __restrict__ outb) {
  __shared__ float red[256];
  int t = blockIdx.x, tid = threadIdx.x;
  float v[3]; float ls = 0.f, lq = 0.f;
  #pragma unroll
  for (int e = 0; e < 3; ++e) {
    int hh = e * 256 + tid;
    float val = res[(size_t)t * Hd + hh] + y[(size_t)t * Hd + hh];
    v[e] = val; ls += val; lq += val * val;
  }
  float sum = block_sum(ls, red);
  float sq  = block_sum(lq, red);
  float mean = sum / Hd;
  float var  = sq / Hd - mean * mean;
  float inv  = rsqrtf(var + 1e-12f);
  #pragma unroll
  for (int e = 0; e < 3; ++e) {
    int hh = e * 256 + tid;
    float o = (v[e] - mean) * inv * g[hh] + bt[hh];
    out[(size_t)t * Hd + hh]  = o;
    outb[(size_t)t * Hd + hh] = f2bf(o);
  }
}

// ---------------------------------------------------------------------------
// Core WMMA GEMM, LDS-free: C = act(A @ Bt^T + bias)
//   A  bf16 [M][K] row-major, Bt bf16 [N][K] row-major (pre-transposed).
// Block = 256 (8 waves, 2x4), block tile 128x256, wave tile 64x64:
// 16 WMMAs per K-step against 8 fragments (32 FLOP per cache byte).
// B fragments are loaded per-column right before use to keep peak VGPR
// liveness ~= 128 acc + 32 A-frag + 8 B-frag, safely under 256.
// All dims are multiples of the tile -> no guards anywhere.
// ---------------------------------------------------------------------------
template <int ACT>
__global__ __launch_bounds__(256) void gemm_bf16_kernel(
    const __bf16* __restrict__ A, const __bf16* __restrict__ Bt,
    const float* __restrict__ bias,
    float* __restrict__ Cf, __bf16* __restrict__ Cb,
    int ldc, int Kd) {
  const int tid  = threadIdx.x;
  const int lane = tid & 31, wave = tid >> 5;
  const int wm = wave >> 2, wn = wave & 3;
  const int half = lane >> 4, l16 = lane & 15;
  const int m0 = blockIdx.y * 128, n0 = blockIdx.x * 256;

  const __bf16* arow[4];
  const __bf16* brow[4];
  #pragma unroll
  for (int i = 0; i < 4; ++i)
    arow[i] = A + (size_t)(m0 + wm * 64 + i * 16 + l16) * Kd;
  #pragma unroll
  for (int j = 0; j < 4; ++j)
    brow[j] = Bt + (size_t)(n0 + wn * 64 + j * 16 + l16) * Kd;

  v8f acc[4][4] = {};
  for (int k0 = 0; k0 < Kd; k0 += 32) {
    v16bf af[4];
    #pragma unroll
    for (int i = 0; i < 4; ++i) af[i] = load_a_frag(arow[i] + k0, half);
    #pragma unroll
    for (int j = 0; j < 4; ++j) {
      v16bf bf_ = load_b_frag(brow[j] + k0, half);
      #pragma unroll
      for (int i = 0; i < 4; ++i)
        acc[i][j] = wmma_bf16(af[i], bf_, acc[i][j]);
    }
  }
  #pragma unroll
  for (int i = 0; i < 4; ++i) {
    int rowb = m0 + wm * 64 + i * 16 + half * 8;   // C layout: vgpr r -> M=half*8+r
    #pragma unroll
    for (int j = 0; j < 4; ++j) {
      int col = n0 + wn * 64 + j * 16 + l16;
      float bv = bias[col];
      #pragma unroll
      for (int r = 0; r < 8; ++r) {
        float v = acc[i][j][r] + bv;
        if (ACT == 1) v = gelu_f(v);
        size_t o = (size_t)(rowb + r) * ldc + col;
        if (Cf) Cf[o] = v;
        if (Cb) Cb[o] = f2bf(v);
      }
    }
  }
}

// ---------------------------------------------------------------------------
// Attention scores: S = (Q Kt)/sqrt(DH) + visibility bias  (f32 out)
// grid = (S/128, S/64, B*NH); Q,K bf16 [M][H], head-sliced rows.
// ---------------------------------------------------------------------------
__global__ __launch_bounds__(256) void attn_scores_kernel(
    const __bf16* __restrict__ q, const __bf16* __restrict__ kmat,
    const int* __restrict__ sepA, float* __restrict__ scores) {
  const int tid  = threadIdx.x;
  const int lane = tid & 31, wave = tid >> 5;
  const int wm = wave >> 2, wn = wave & 3;
  const int half = lane >> 4, l16 = lane & 15;
  const int z = blockIdx.z, b = z / NHd, h = z % NHd;
  const int m0 = blockIdx.y * 64, n0 = blockIdx.x * 128;
  const __bf16* Qp = q    + (size_t)b * Sq * Hd + h * DHd;
  const __bf16* Kp = kmat + (size_t)b * Sq * Hd + h * DHd;

  const __bf16* arow[2];
  const __bf16* brow[2];
  #pragma unroll
  for (int i = 0; i < 2; ++i)
    arow[i] = Qp + (size_t)(m0 + wm * 32 + i * 16 + l16) * Hd;
  #pragma unroll
  for (int j = 0; j < 2; ++j)
    brow[j] = Kp + (size_t)(n0 + wn * 32 + j * 16 + l16) * Hd;  // B[d][j]=K[j][d]

  v8f acc[2][2] = {};
  #pragma unroll
  for (int k0 = 0; k0 < DHd; k0 += 32) {
    v16bf af[2], bf_[2];
    #pragma unroll
    for (int i = 0; i < 2; ++i) af[i]  = load_a_frag(arow[i] + k0, half);
    #pragma unroll
    for (int j = 0; j < 2; ++j) bf_[j] = load_b_frag(brow[j] + k0, half);
    #pragma unroll
    for (int i = 0; i < 2; ++i)
      #pragma unroll
      for (int j = 0; j < 2; ++j)
        acc[i][j] = wmma_bf16(af[i], bf_[j], acc[i][j]);
  }
  const int sb = sepA[b];
  float* outz = scores + (size_t)z * Sq * Sq;
  #pragma unroll
  for (int i = 0; i < 2; ++i) {
    int rowb = m0 + wm * 32 + i * 16 + half * 8;
    #pragma unroll
    for (int j = 0; j < 2; ++j) {
      int col = n0 + wn * 32 + j * 16 + l16;
      #pragma unroll
      for (int r = 0; r < 8; ++r) {
        int irow = rowb + r;
        bool vis = (sb < 0) || (col <= sb) || ((irow > sb) && (col <= irow));
        outz[(size_t)irow * Sq + col] =
            acc[i][j][r] * 0.125f + (vis ? 0.0f : -1.0e9f);
      }
    }
  }
}

// ---------------------------------------------------------------------------
// Row softmax over S=512: f32 scores in, bf16 probs out.
// ---------------------------------------------------------------------------
__global__ __launch_bounds__(256) void softmax_kernel(
    const float* __restrict__ sc, __bf16* __restrict__ pb) {
  __shared__ float red[256];
  const size_t row = blockIdx.x;
  const float* p = sc + row * Sq;
  __bf16* o = pb + row * Sq;
  int tid = threadIdx.x;
  float a0 = p[tid], a1 = p[tid + 256];
  red[tid] = fmaxf(a0, a1); __syncthreads();
  for (int s = 128; s > 0; s >>= 1) {
    if (tid < s) red[tid] = fmaxf(red[tid], red[tid + s]);
    __syncthreads();
  }
  float mx = red[0]; __syncthreads();
  float e0 = __expf(a0 - mx), e1 = __expf(a1 - mx);
  red[tid] = e0 + e1; __syncthreads();
  for (int s = 128; s > 0; s >>= 1) {
    if (tid < s) red[tid] += red[tid + s];
    __syncthreads();
  }
  float inv = 1.0f / red[0];
  o[tid]       = f2bf(e0 * inv);
  o[tid + 256] = f2bf(e1 * inv);
}

// ---------------------------------------------------------------------------
// ctx = P @ V per (b,head). Tile 128x64, 8 waves (4x2). bf16 out (head-sliced).
// probs bf16 [z][S][S]; vt bf16 [z*DH + d][S]. grid = (1, S/128, B*NH)
// ---------------------------------------------------------------------------
__global__ __launch_bounds__(256) void attn_ctx_kernel(
    const __bf16* __restrict__ probs, const __bf16* __restrict__ vt,
    __bf16* __restrict__ ctxb) {
  const int tid  = threadIdx.x;
  const int lane = tid & 31, wave = tid >> 5;
  const int wm = wave >> 1, wn = wave & 1;
  const int half = lane >> 4, l16 = lane & 15;
  const int z = blockIdx.z, b = z / NHd, h = z % NHd;
  const int m0 = blockIdx.y * 128;
  const __bf16* Pp = probs + (size_t)z * Sq * Sq;
  const __bf16* Vt = vt + (size_t)z * DHd * Sq;

  const __bf16* arow[2];
  const __bf16* brow[2];
  #pragma unroll
  for (int i = 0; i < 2; ++i)
    arow[i] = Pp + (size_t)(m0 + wm * 32 + i * 16 + l16) * Sq;
  #pragma unroll
  for (int j = 0; j < 2; ++j)
    brow[j] = Vt + (size_t)(wn * 32 + j * 16 + l16) * Sq;

  v8f acc[2][2] = {};
  #pragma unroll 2
  for (int k0 = 0; k0 < Sq; k0 += 32) {
    v16bf af[2], bf_[2];
    #pragma unroll
    for (int i = 0; i < 2; ++i) af[i]  = load_a_frag(arow[i] + k0, half);
    #pragma unroll
    for (int j = 0; j < 2; ++j) bf_[j] = load_b_frag(brow[j] + k0, half);
    #pragma unroll
    for (int i = 0; i < 2; ++i)
      #pragma unroll
      for (int j = 0; j < 2; ++j)
        acc[i][j] = wmma_bf16(af[i], bf_[j], acc[i][j]);
  }
  #pragma unroll
  for (int i = 0; i < 2; ++i) {
    int rowb = m0 + wm * 32 + i * 16 + half * 8;
    #pragma unroll
    for (int j = 0; j < 2; ++j) {
      int col = wn * 32 + j * 16 + l16;            // d in [0,64)
      #pragma unroll
      for (int r = 0; r < 8; ++r) {
        int irow = rowb + r;
        ctxb[((size_t)b * Sq + irow) * Hd + h * DHd + col] = f2bf(acc[i][j][r]);
      }
    }
  }
}

// ---------------------------------------------------------------------------
// Classifier + CE, logits never materialized. xb bf16 [M][H], Wt bf16
// [Vpad][H] (zero-padded), online logsumexp per row, label logit captured.
// grid = M/64 blocks of 256.
// ---------------------------------------------------------------------------
__global__ __launch_bounds__(256) void cls_ce_kernel(
    const __bf16* __restrict__ xb, const __bf16* __restrict__ Wt,
    const float* __restrict__ bias, const int* __restrict__ labels,
    const int* __restrict__ firstA, float* __restrict__ cem) {
  __shared__ float mxs[4][64], sms[4][64], lab[64];
  __shared__ int labs[64];
  const int tid  = threadIdx.x;
  const int lane = tid & 31, wave = tid >> 5;
  const int wm = wave >> 2, wn = wave & 3;
  const int half = lane >> 4, l16 = lane & 15;
  const int m0 = blockIdx.x * 64;

  if (tid < 64) { lab[tid] = 0.0f; labs[tid] = labels[m0 + tid]; }
  __syncthreads();

  const __bf16* arow[2];
  #pragma unroll
  for (int i = 0; i < 2; ++i)
    arow[i] = xb + (size_t)(m0 + wm * 32 + i * 16 + l16) * Hd;

  float mx[2][8], sm[2][8];
  #pragma unroll
  for (int i = 0; i < 2; ++i)
    #pragma unroll
    for (int r = 0; r < 8; ++r) { mx[i][r] = -3.0e38f; sm[i][r] = 0.0f; }

  for (int n0 = 0; n0 < Vpad; n0 += 128) {
    const __bf16* brow[2];
    #pragma unroll
    for (int j = 0; j < 2; ++j)
      brow[j] = Wt + (size_t)(n0 + wn * 32 + j * 16 + l16) * Hd;

    v8f acc[2][2] = {};
    for (int k0 = 0; k0 < Hd; k0 += 32) {
      v16bf af[2], bf_[2];
      #pragma unroll
      for (int i = 0; i < 2; ++i) af[i]  = load_a_frag(arow[i] + k0, half);
      #pragma unroll
      for (int j = 0; j < 2; ++j) bf_[j] = load_b_frag(brow[j] + k0, half);
      #pragma unroll
      for (int i = 0; i < 2; ++i)
        #pragma unroll
        for (int j = 0; j < 2; ++j)
          acc[i][j] = wmma_bf16(af[i], bf_[j], acc[i][j]);
    }
    // online logsumexp update for this 128-col tile (pad cols -> -inf)
    #pragma unroll
    for (int i = 0; i < 2; ++i) {
      #pragma unroll
      for (int j = 0; j < 2; ++j) {
        int col = n0 + wn * 32 + j * 16 + l16;
        bool inr = col < Vc;
        float bv = inr ? bias[col] : 0.0f;
        #pragma unroll
        for (int r = 0; r < 8; ++r) {
          int lrow = wm * 32 + i * 16 + half * 8 + r;
          float v  = inr ? (acc[i][j][r] + bv) : -3.0e38f;
          float nm = fmaxf(mx[i][r], v);
          sm[i][r] = sm[i][r] * __expf(mx[i][r] - nm) + __expf(v - nm);
          mx[i][r] = nm;
          if (inr && col == labs[lrow]) lab[lrow] = v;
        }
      }
    }
  }
  // cross-lane combine within 16-lane halves (rows striped across n-lanes)
  #pragma unroll
  for (int i = 0; i < 2; ++i) {
    #pragma unroll
    for (int r = 0; r < 8; ++r) {
      float m = mx[i][r], s0 = sm[i][r];
      #pragma unroll
      for (int off = 1; off < 16; off <<= 1) {
        float om = __shfl_xor(m,  off, 32);
        float os = __shfl_xor(s0, off, 32);
        float nm = fmaxf(m, om);
        s0 = s0 * __expf(m - nm) + os * __expf(om - nm);
        m = nm;
      }
      if (l16 == 0) {
        int lrow = wm * 32 + i * 16 + half * 8 + r;
        mxs[wn][lrow] = m; sms[wn][lrow] = s0;
      }
    }
  }
  __syncthreads();
  if (tid < 64) {
    float m = mxs[0][tid];
    #pragma unroll
    for (int w = 1; w < 4; ++w) m = fmaxf(m, mxs[w][tid]);
    float s0 = 0.0f;
    #pragma unroll
    for (int w = 0; w < 4; ++w) s0 += sms[w][tid] * __expf(mxs[w][tid] - m);
    float lse = m + logf(s0);
    float ce  = lse - lab[tid];
    int t = m0 + tid;
    int b = t >> 9, i = t & (Sq - 1);
    int f = firstA[b];
    float msk = (f < 0 || i > f) ? 1.0f : 0.0f;
    cem[t] = ce * msk;
  }
}

// ---------------------------------------------------------------------------
// Final masked-mean reduction -> scalar loss
// ---------------------------------------------------------------------------
__global__ __launch_bounds__(256) void reduce_kernel(
    const float* __restrict__ cem, const int* __restrict__ firstA,
    float* __restrict__ out) {
  __shared__ float red[256];
  int tid = threadIdx.x;
  float accum = 0.0f;
  for (int b = 0; b < Bz; ++b) {
    float s = 0.0f;
    for (int i = tid; i < Sq; i += 256) s += cem[b * Sq + i];
    s = block_sum(s, red);
    int f = firstA[b];
    float cnt = (f < 0) ? (float)Sq : (float)(Sq - 1 - f);
    accum += s / cnt;
  }
  if (tid == 0) out[0] = accum / (float)Bz;
}

// ---------------------------------------------------------------------------
// Host orchestration
// ---------------------------------------------------------------------------
extern "C" void kernel_launch(void* const* d_in, const int* in_sizes, int n_in,
                              void* d_out, int out_size, void* d_ws, size_t ws_size,
                              hipStream_t stream) {
  (void)in_sizes; (void)n_in; (void)out_size; (void)ws_size;
  const int*   input_ids = (const int*)  d_in[0];
  const int*   labels    = (const int*)  d_in[1];
  const float* word_emb  = (const float*)d_in[2];
  const float* pos_emb   = (const float*)d_in[3];
  const float* type_emb  = (const float*)d_in[4];
  const float* emb_g     = (const float*)d_in[5];
  const float* emb_b     = (const float*)d_in[6];
  const float* Wq        = (const float*)d_in[7];
  const float* bq        = (const float*)d_in[8];
  const float* Wk        = (const float*)d_in[9];
  const float* bk        = (const float*)d_in[10];
  const float* Wv        = (const float*)d_in[11];
  const float* bv        = (const float*)d_in[12];
  const float* Wo        = (const float*)d_in[13];
  const float* bo        = (const float*)d_in[14];
  const float* ln1_g     = (const float*)d_in[15];
  const float* ln1_b     = (const float*)d_in[16];
  const float* W1        = (const float*)d_in[17];
  const float* b1        = (const float*)d_in[18];
  const float* W2        = (const float*)d_in[19];
  const float* b2        = (const float*)d_in[20];
  const float* ln2_g     = (const float*)d_in[21];
  const float* ln2_b     = (const float*)d_in[22];
  const float* cls_W     = (const float*)d_in[23];
  const float* cls_b     = (const float*)d_in[24];
  float* out = (float*)d_out;

  // workspace carve (bytes, 256B aligned)
  char* wsp = (char*)d_ws;
  size_t off = 0;
  auto alloc = [&](size_t bytes) -> void* {
    void* r = wsp + off;
    off += (bytes + 255) & ~(size_t)255;
    return r;
  };
  const size_t MH  = (size_t)Mtok * Hd;
  const size_t SS  = (size_t)Bz * NHd * Sq * Sq;
  const size_t LW  = (size_t)4 * Hd * Hd + (size_t)2 * Hd * Ff;  // per-layer Wt elems

  __bf16* wts    = (__bf16*)alloc(LW * Ly * 2);            // all layer weights, bf16 T
  __bf16* clsWt  = (__bf16*)alloc((size_t)Vpad * Hd * 2);  // padded cls weight, bf16 T
  float*  x      = (float*) alloc(MH * 4);
  __bf16* xbuf   = (__bf16*)alloc(MH * 2);
  __bf16* qbb    = (__bf16*)alloc(MH * 2);
  __bf16* kbb    = (__bf16*)alloc(MH * 2);
  __bf16* vbb    = (__bf16*)alloc(MH * 2);
  __bf16* vt     = (__bf16*)alloc(MH * 2);
  float*  scores = (float*) alloc(SS * 4);
  __bf16* probsb = (__bf16*)alloc(SS * 2);
  __bf16* ctxb   = (__bf16*)alloc(MH * 2);
  float*  tmp1   = (float*) alloc(MH * 4);
  float*  h1     = (float*) alloc(MH * 4);
  __bf16* h1b    = (__bf16*)alloc(MH * 2);
  __bf16* midb   = (__bf16*)alloc((size_t)Mtok * Ff * 2);
  float*  cem    = (float*) alloc((size_t)Mtok * 4);
  int*    sepA   = (int*)   alloc(Bz * 4);
  int*    firstA = (int*)   alloc(Bz * 4);

  dim3 tb(32, 8);
  auto wconv = [&](const float* W, __bf16* Wt, int Kd, int Nd, int Npad) {
    wconv_kernel<<<dim3(Npad / 32, Kd / 32), tb, 0, stream>>>(W, Wt, Kd, Nd, Npad);
  };

  // --- one-time (per launch) weight convert+transpose ---
  for (int l = 0; l < Ly; ++l) {
    __bf16* base = wts + LW * l;
    __bf16* wqt = base;
    __bf16* wkt = wqt + (size_t)Hd * Hd;
    __bf16* wvt = wkt + (size_t)Hd * Hd;
    __bf16* wot = wvt + (size_t)Hd * Hd;
    __bf16* w1t = wot + (size_t)Hd * Hd;          // [F][H]
    __bf16* w2t = w1t + (size_t)Ff * Hd;          // [H][F]
    wconv(Wq + (size_t)l * Hd * Hd, wqt, Hd, Hd, Hd);
    wconv(Wk + (size_t)l * Hd * Hd, wkt, Hd, Hd, Hd);
    wconv(Wv + (size_t)l * Hd * Hd, wvt, Hd, Hd, Hd);
    wconv(Wo + (size_t)l * Hd * Hd, wot, Hd, Hd, Hd);
    wconv(W1 + (size_t)l * Hd * Ff, w1t, Hd, Ff, Ff);
    wconv(W2 + (size_t)l * Ff * Hd, w2t, Ff, Hd, Hd);
  }
  wconv(cls_W, clsWt, Hd, Vc, Vpad);

  sep_kernel<<<1, 32, 0, stream>>>(input_ids, sepA, firstA);
  embed_ln_kernel<<<Mtok, 256, 0, stream>>>(input_ids, word_emb, pos_emb,
                                            type_emb, emb_g, emb_b, x, xbuf);

  auto gemm = [&](const __bf16* A, const __bf16* Bt, const float* bias,
                  float* Cf, __bf16* Cb, int Nd, int Kd, int act) {
    dim3 grid(Nd / 256, Mtok / 128, 1);
    if (act == 0)
      gemm_bf16_kernel<0><<<grid, 256, 0, stream>>>(A, Bt, bias, Cf, Cb, Nd, Kd);
    else
      gemm_bf16_kernel<1><<<grid, 256, 0, stream>>>(A, Bt, bias, Cf, Cb, Nd, Kd);
  };

  for (int l = 0; l < Ly; ++l) {
    __bf16* base = wts + LW * l;
    __bf16* wqt = base;
    __bf16* wkt = wqt + (size_t)Hd * Hd;
    __bf16* wvt = wkt + (size_t)Hd * Hd;
    __bf16* wot = wvt + (size_t)Hd * Hd;
    __bf16* w1t = wot + (size_t)Hd * Hd;
    __bf16* w2t = w1t + (size_t)Ff * Hd;

    gemm(xbuf, wqt, bq + (size_t)l * Hd, nullptr, qbb, Hd, Hd, 0);
    gemm(xbuf, wkt, bk + (size_t)l * Hd, nullptr, kbb, Hd, Hd, 0);
    gemm(xbuf, wvt, bv + (size_t)l * Hd, nullptr, vbb, Hd, Hd, 0);
    transpose_v_kernel<<<dim3(DHd / 32, Sq / 32, Bz * NHd), tb, 0, stream>>>(vbb, vt);

    attn_scores_kernel<<<dim3(Sq / 128, Sq / 64, Bz * NHd), 256, 0, stream>>>(
        qbb, kbb, sepA, scores);
    softmax_kernel<<<Bz * NHd * Sq, 256, 0, stream>>>(scores, probsb);
    attn_ctx_kernel<<<dim3(1, Sq / 128, Bz * NHd), 256, 0, stream>>>(
        probsb, vt, ctxb);

    gemm(ctxb, wot, bo + (size_t)l * Hd, tmp1, nullptr, Hd, Hd, 0);
    add_res_ln_kernel<<<Mtok, 256, 0, stream>>>(
        x, tmp1, ln1_g + (size_t)l * Hd, ln1_b + (size_t)l * Hd, h1, h1b);

    gemm(h1b, w1t, b1 + (size_t)l * Ff, nullptr, midb, Ff, Hd, 1);
    gemm(midb, w2t, b2 + (size_t)l * Hd, tmp1, nullptr, Hd, Ff, 0);
    add_res_ln_kernel<<<Mtok, 256, 0, stream>>>(
        h1, tmp1, ln2_g + (size_t)l * Hd, ln2_b + (size_t)l * Hd, x, xbuf);
  }

  cls_ce_kernel<<<Mtok / 64, 256, 0, stream>>>(xbuf, clsWt, cls_b, labels,
                                               firstA, cem);
  reduce_kernel<<<1, 256, 0, stream>>>(cem, firstA, out);
}
